// MHSelfAttention_37847251813112
// MI455X (gfx1250) — matrix operand order repaired and checked by reading, not compiled, not run on hardware
//
#include <hip/hip_runtime.h>
#include <hip/hip_bf16.h>

// ---------------------------------------------------------------------------
// Enformer-style MHSA with relative positions on gfx1250 (MI455X).
// f16 WMMA (v_wmma_f32_16x16x32_f16) everywhere, f32 accumulate.
// Flash-style online softmax; relative_shift done as index remap on a
// 16x48 WMMA band with width-16 shuffles (no (B,H,L,2L-1) tensor).
// Round 2: double-buffered GEMM k-loop, hoisted V fragments in attention
// so WMMAs overlap outstanding loads instead of s_wait_loadcnt 0 drains.
// ---------------------------------------------------------------------------

typedef __attribute__((ext_vector_type(16))) _Float16 v16h;
typedef __attribute__((ext_vector_type(8)))  _Float16 half8;
typedef __attribute__((ext_vector_type(8)))  float    v8f;

#define BB   4
#define LL   1536
#define DD   768
#define HH   8
#define DQKc 64
#define DVc  96
#define NPFc 96
#define PP   3071      // 2L-1
#define PPAD 3072
#define NBc  16        // basis per feature fn (NPF/6)

__device__ __forceinline__ v8f v8f_zero() {
  v8f z = {0.f, 0.f, 0.f, 0.f, 0.f, 0.f, 0.f, 0.f};
  return z;
}

__device__ __forceinline__ v8f wmma_f16(v16h a, v16h b, v8f c) {
  // emits v_wmma_f32_16x16x32_f16
  return __builtin_amdgcn_wmma_f32_16x16x32_f16(false, a, false, b, (short)0, c,
                                                false, false);
}

// A fragment (16x32 f16): lane m = lane&15, group g = lane>>4.
// halves: [m][8g .. 8g+7] then [m][16+8g .. 16+8g+7]    (base = row ptr + k0)
__device__ __forceinline__ v16h load_a_frag(const _Float16* base, int g) {
  union { v16h v; half8 h[2]; } u;
  u.h[0] = *(const half8*)(base + 8 * g);
  u.h[1] = *(const half8*)(base + 16 + 8 * g);
  return u.v;
}

// B fragment (32x16 f16): lane n = lane&15 holds k-halves [8g..8g+7] then [16+8g..16+8g+7]
// (base = column-row ptr + k0; operand matrix stored so k-dim is contiguous)
__device__ __forceinline__ v16h load_b_frag(const _Float16* base, int g) {
  union { v16h v; half8 h[2]; } u;
  const _Float16* p = base + 8 * g;
  u.h[0] = *(const half8*)(p);
  u.h[1] = *(const half8*)(p + 16);
  return u.v;
}

// ---------------------------------------------------------------------------
// Converters
// ---------------------------------------------------------------------------
__global__ void cvt_f16_kernel(const float* __restrict__ src,
                               _Float16* __restrict__ dst, int n) {
  int i = blockIdx.x * blockDim.x + threadIdx.x;
  if (i < n) dst[i] = (_Float16)src[i];
}

// dst[n*K + k] = src[k*N + n] * scale   (transpose so k-dim is contiguous)
__global__ void cvt_f16_t_kernel(const float* __restrict__ src,
                                 _Float16* __restrict__ dst, int K, int N,
                                 float scale) {
  int i = blockIdx.x * blockDim.x + threadIdx.x;
  if (i >= K * N) return;
  int k = i / N, n = i % N;
  dst[(size_t)n * K + k] = (_Float16)(src[i] * scale);
}

// ---------------------------------------------------------------------------
// Positional features
// ---------------------------------------------------------------------------
__global__ void init_gmax_kernel(unsigned* gmax) {
  if (threadIdx.x < NBc) gmax[threadIdx.x] = 0u;
}

__global__ void gamma_unnorm_kernel(float* __restrict__ gbuf,
                                    unsigned* __restrict__ gmax) {
  int i = blockIdx.x * blockDim.x + threadIdx.x;
  if (i >= PP * NBc) return;
  int p = i >> 4, f = i & 15;
  float pos = (float)p - (float)(LL - 1);
  float x = fabsf(pos);
  float mf = (float)(f + 1);
  float conc = 4.f * mf * mf;           // ((96*(f+1))/48)^2
  float rate = mf / 24.f;               // 96*(f+1)/48^2
  float log_norm = lgammaf(conc) - conc * logf(rate);
  float lu = (x > 0.f) ? (conc - 1.f) * logf(x) - rate * x : -__builtin_inff();
  float prob = __expf(lu - log_norm) + 1e-8f;
  gbuf[i] = prob;
  atomicMax(&gmax[f], __float_as_uint(prob));  // all values > 0
}

__global__ void pe_assemble_kernel(const float* __restrict__ gbuf,
                                   const unsigned* __restrict__ gmax,
                                   _Float16* __restrict__ pe) {
  int i = blockIdx.x * blockDim.x + threadIdx.x;
  if (i >= PPAD * 48) return;
  int p = i / 48, c = i % 48;
  if (p >= PP) {  // zero pad row
    pe[(size_t)p * NPFc + c] = (_Float16)0.f;
    pe[(size_t)p * NPFc + 48 + c] = (_Float16)0.f;
    return;
  }
  float pos = (float)p - (float)(LL - 1);
  float x = fabsf(pos);
  float v;
  if (c < 16) {  // exponential
    const float maxrange = 10.584962500721156f;  // log2(1536)
    float t = 3.f + (float)c * ((maxrange - 3.f) / 15.f);
    float hl = exp2f(t);
    v = exp2f(-x / hl);
  } else if (c < 32) {  // central mask
    int f = c - 16;
    float cw = exp2f((float)(f + 1)) - 1.f;
    v = (cw > x) ? 1.f : 0.f;
  } else {  // gamma, normalized by per-feature max
    int f = c - 32;
    v = gbuf[p * NBc + f] / __uint_as_float(gmax[f]);
  }
  float s = (pos > 0.f) ? 1.f : ((pos < 0.f) ? -1.f : 0.f);
  pe[(size_t)p * NPFc + c] = (_Float16)v;
  pe[(size_t)p * NPFc + 48 + c] = (_Float16)(v * s);
}

// ---------------------------------------------------------------------------
// Generic WMMA GEMM: C(MxN) = A(MxK,f16 rowmajor) @ Wt(NxK,f16 k-contig)
// block = 256 threads (8 waves); block tile 32x256; wave tile 16x64.
// k-loop is software double-buffered: iteration k+1's 5 fragments are issued
// before iteration k's 4 WMMAs so the in-order LOADcnt lets WMMA run with
// loads outstanding (partial s_wait_loadcnt instead of full drains).
// store modes:
//   0: f16 row-major out0
//   1: f16 head-split   out0[((b*H+h)*Lp+l)*dh+j]
//   2: f16 V-transposed out0[((b*H+h)*dh+j)*Lp+l]
//   3: f32 + bias0 -> out0[row*N+col]  (final output)
//   4: dual f16 head-split with biases: out0 = acc+bias0, out1 = acc+bias1
// ---------------------------------------------------------------------------
__global__ __launch_bounds__(256) void gemm_wmma_kernel(
    const _Float16* __restrict__ A, const _Float16* __restrict__ Wt,
    void* __restrict__ out0, void* __restrict__ out1,
    const float* __restrict__ bias0, const float* __restrict__ bias1,
    int M, int N, int K, int Lp, int dh, int mode) {
  int lane = threadIdx.x & 31, wid = threadIdx.x >> 5;
  int wm = wid >> 2, wn = wid & 3;
  int m0 = blockIdx.y * 32 + wm * 16;
  int n0 = blockIdx.x * 256 + wn * 64;
  int ml = lane & 15, g = lane >> 4;

  const _Float16* arow = A + (size_t)(m0 + ml) * K;
  const _Float16* b0 = Wt + (size_t)(n0 + 0 + ml) * K;
  const _Float16* b1 = Wt + (size_t)(n0 + 16 + ml) * K;
  const _Float16* b2 = Wt + (size_t)(n0 + 32 + ml) * K;
  const _Float16* b3 = Wt + (size_t)(n0 + 48 + ml) * K;

  v8f c0 = v8f_zero(), c1 = v8f_zero(), c2 = v8f_zero(), c3 = v8f_zero();

  // prologue: fragments for k0 = 0
  v16h a  = load_a_frag(arow, g);
  v16h f0 = load_b_frag(b0, g);
  v16h f1 = load_b_frag(b1, g);
  v16h f2 = load_b_frag(b2, g);
  v16h f3 = load_b_frag(b3, g);

  for (int k0 = 0; k0 < K; k0 += 32) {
    // issue next iteration's loads first (clamped on the last step so the
    // loop stays branchless and EXEC stays all-ones for WMMA)
    int kn = (k0 + 32 < K) ? (k0 + 32) : k0;
    __builtin_prefetch(arow + kn + 32, 0, 1);
    v16h an  = load_a_frag(arow + kn, g);
    v16h fn0 = load_b_frag(b0 + kn, g);
    v16h fn1 = load_b_frag(b1 + kn, g);
    v16h fn2 = load_b_frag(b2 + kn, g);
    v16h fn3 = load_b_frag(b3 + kn, g);
    c0 = wmma_f16(a, f0, c0);
    c1 = wmma_f16(a, f1, c1);
    c2 = wmma_f16(a, f2, c2);
    c3 = wmma_f16(a, f3, c3);
    a = an; f0 = fn0; f1 = fn1; f2 = fn2; f3 = fn3;
  }

  v8f cc[4] = {c0, c1, c2, c3};
#pragma unroll
  for (int t = 0; t < 4; ++t) {
#pragma unroll
    for (int r = 0; r < 8; ++r) {
      int row = m0 + r + 8 * g;
      int col = n0 + 16 * t + ml;
      float v = cc[t][r];
      if (mode == 0) {
        ((_Float16*)out0)[(size_t)row * N + col] = (_Float16)v;
      } else if (mode == 3) {
        ((float*)out0)[(size_t)row * N + col] = v + bias0[col];
      } else {
        int b = row / Lp, l = row % Lp, h = col / dh, j = col % dh;
        if (mode == 1) {
          ((_Float16*)out0)[(((size_t)b * HH + h) * Lp + l) * dh + j] =
              (_Float16)v;
        } else if (mode == 2) {
          ((_Float16*)out0)[(((size_t)b * HH + h) * dh + j) * Lp + l] =
              (_Float16)v;
        } else {  // mode 4
          size_t idx = (((size_t)b * HH + h) * Lp + l) * dh + j;
          ((_Float16*)out0)[idx] = (_Float16)(v + bias0[col]);
          ((_Float16*)out1)[idx] = (_Float16)(v + bias1[col]);
        }
      }
    }
  }
}

// ---------------------------------------------------------------------------
// Attention: flash-style online softmax, per-wave 16 query rows.
// grid = (L/64, B*H), block = 128 (4 waves). S never hits memory.
// rel term: 16x48 band R = Qr @ rK[jb0..jb0+47]^T ; shifted element
// (i,k) = R[i, (k-k0)-(i-i0)+15 (+16*sub)] gathered with width-16 shuffles.
// V fragments are issued BEFORE the softmax VALU section so the long
// shuffle/exp sequence plus the LDS bounce hide their latency.
// ---------------------------------------------------------------------------
__global__ __launch_bounds__(128) void attn_wmma_kernel(
    const _Float16* __restrict__ Qw, const _Float16* __restrict__ Qr,
    const _Float16* __restrict__ Kh, const _Float16* __restrict__ Vt,
    const _Float16* __restrict__ rK, _Float16* __restrict__ attn) {
  __shared__ _Float16 plds[4][16 * 32];  // P-tile C->A layout bounce, per wave
  int lane = threadIdx.x & 31, wid = threadIdx.x >> 5;
  int ml = lane & 15, g = lane >> 4;
  int bh = blockIdx.y, b = bh >> 3, h = bh & 7;
  int i0 = blockIdx.x * 64 + wid * 16;

  const _Float16* qwrow = Qw + ((size_t)bh * LL + i0 + ml) * DQKc;
  const _Float16* qrrow = Qr + ((size_t)bh * LL + i0 + ml) * DQKc;
  v16h aqw[2], aqr[2];
  aqw[0] = load_a_frag(qwrow, g);
  aqw[1] = load_a_frag(qwrow + 32, g);
  aqr[0] = load_a_frag(qrrow, g);
  aqr[1] = load_a_frag(qrrow + 32, g);

  const _Float16* kbase = Kh + (size_t)bh * LL * DQKc;
  const _Float16* vbase = Vt + (size_t)bh * DVc * LL;
  const _Float16* rbase = rK + (size_t)h * PPAD * DQKc;
  _Float16* pl = &plds[wid][0];

  v8f o[6];
  float mr[8], ls[8];
#pragma unroll
  for (int t = 0; t < 6; ++t) o[t] = v8f_zero();
#pragma unroll
  for (int r = 0; r < 8; ++r) { mr[r] = -3.0e38f; ls[r] = 0.f; }

  for (int it = 0; it < LL / 32; ++it) {
    int kk0 = it * 32;
    // ---- content scores: S = Qw @ K^T, 16 x 32 (two 16x16 C-frags) ----
    v8f s0 = v8f_zero(), s1 = v8f_zero();
#pragma unroll
    for (int s = 0; s < 2; ++s) {
      v16h bk0 = load_b_frag(kbase + (size_t)(kk0 + ml) * DQKc + 32 * s, g);
      v16h bk1 = load_b_frag(kbase + (size_t)(kk0 + 16 + ml) * DQKc + 32 * s, g);
      s0 = wmma_f16(aqw[s], bk0, s0);
      s1 = wmma_f16(aqw[s], bk1, s1);
    }
    // ---- rel band: R = Qr @ rK[jb0 .. jb0+47]^T (3 C-frags) ----
    int jb0 = kk0 - i0 + (LL - 16);  // (k0 - i0) + (L-1) - 15 >= 0
    v8f r0 = v8f_zero(), r1 = v8f_zero(), r2 = v8f_zero();
#pragma unroll
    for (int s = 0; s < 2; ++s) {
      r0 = wmma_f16(aqr[s],
                    load_b_frag(rbase + (size_t)(jb0 + ml) * DQKc + 32 * s, g),
                    r0);
      r1 = wmma_f16(
          aqr[s],
          load_b_frag(rbase + (size_t)(jb0 + 16 + ml) * DQKc + 32 * s, g), r1);
      r2 = wmma_f16(
          aqr[s],
          load_b_frag(rbase + (size_t)(jb0 + 32 + ml) * DQKc + 32 * s, g), r2);
    }
    // ---- issue V fragments now; softmax VALU below hides their latency ----
    v16h bv[6];
#pragma unroll
    for (int t = 0; t < 6; ++t)
      bv[t] = load_b_frag(vbase + (size_t)(16 * t + ml) * LL + kk0, g);
    // prefetch next iteration's K / rK rows
    __builtin_prefetch(kbase + (size_t)(kk0 + 32 + ml) * DQKc, 0, 1);
    __builtin_prefetch(rbase + (size_t)(jb0 + 32 + ml) * DQKc + 64, 0, 1);
    // ---- relative_shift as gather: logits[m,kk] += R[m, kk - m + 15(+16)] --
#pragma unroll
    for (int r = 0; r < 8; ++r) {
      int m = r + 8 * g;
      int c0 = ml - m + 15;  // in [0, 30]; sub1 adds 16 -> [16, 46]
      int cl = c0 & 15, t0 = c0 >> 4;
      float f0 = __shfl(r0[r], cl, 16);
      float f1 = __shfl(r1[r], cl, 16);
      float f2 = __shfl(r2[r], cl, 16);
      s0[r] += (t0 == 0) ? f0 : f1;
      s1[r] += (t0 == 0) ? f1 : f2;
    }
    // ---- online softmax over the 32 keys ----
#pragma unroll
    for (int r = 0; r < 8; ++r) {
      float x = fmaxf(s0[r], s1[r]);
      x = fmaxf(x, __shfl_xor(x, 1, 16));
      x = fmaxf(x, __shfl_xor(x, 2, 16));
      x = fmaxf(x, __shfl_xor(x, 4, 16));
      x = fmaxf(x, __shfl_xor(x, 8, 16));
      float mn = fmaxf(mr[r], x);
      float al = __expf(mr[r] - mn);
      mr[r] = mn;
      float p0 = __expf(s0[r] - mn);
      float p1 = __expf(s1[r] - mn);
      float rs = p0 + p1;
      rs += __shfl_xor(rs, 1, 16);
      rs += __shfl_xor(rs, 2, 16);
      rs += __shfl_xor(rs, 4, 16);
      rs += __shfl_xor(rs, 8, 16);
      ls[r] = ls[r] * al + rs;
#pragma unroll
      for (int t = 0; t < 6; ++t) o[t][r] *= al;
      int m = r + 8 * g;
      pl[m * 32 + ml] = (_Float16)p0;
      pl[m * 32 + 16 + ml] = (_Float16)p1;
    }
    // wave-private LDS bounce: C-layout P -> A-layout fragment
    asm volatile("s_wait_dscnt 0" ::: "memory");
    v16h pa = load_a_frag(pl + ml * 32, g);
    // ---- O += P @ V ----
#pragma unroll
    for (int t = 0; t < 6; ++t) o[t] = wmma_f16(pa, bv[t], o[t]);
  }
  // ---- epilogue: normalize, store attn as f16 [B][L][H*DV] ----
#pragma unroll
  for (int r = 0; r < 8; ++r) {
    float inv = 1.f / ls[r];
    int m = r + 8 * g;
    size_t rowoff = ((size_t)(b * LL + i0 + m)) * (HH * DVc) + h * DVc;
#pragma unroll
    for (int t = 0; t < 6; ++t)
      attn[rowoff + 16 * t + ml] = (_Float16)(o[t][r] * inv);
  }
}

// ---------------------------------------------------------------------------
// Host launcher
// ---------------------------------------------------------------------------
extern "C" void kernel_launch(void* const* d_in, const int* in_sizes, int n_in,
                              void* d_out, int out_size, void* d_ws,
                              size_t ws_size, hipStream_t stream) {
  (void)in_sizes; (void)n_in; (void)out_size; (void)ws_size;
  const float* x     = (const float*)d_in[0];  // (B,L,D)
  const float* Q_w   = (const float*)d_in[1];  // (D, H*DQK)
  const float* K_w   = (const float*)d_in[2];  // (D, H*DQK)
  const float* V_w   = (const float*)d_in[3];  // (D, H*DV)
  const float* out_w = (const float*)d_in[4];  // (D, H*DV)
  const float* out_b = (const float*)d_in[5];  // (H*DV,)
  const float* relKw = (const float*)d_in[6];  // (NPF, H*DQK)
  const float* rwb   = (const float*)d_in[7];  // (1,H,1,DQK) -> 512
  const float* rrb   = (const float*)d_in[8];  // (1,H,1,DQK) -> 512

  char* ws = (char*)d_ws;
  size_t off = 0;
  auto alloc = [&](size_t bytes) -> char* {
    char* p = ws + off;
    off += (bytes + 255) & ~(size_t)255;
    return p;
  };
  const int M = BB * LL;  // 6144
  _Float16* xh  = (_Float16*)alloc((size_t)M * DD * 2);
  _Float16* wq  = (_Float16*)alloc((size_t)512 * DD * 2);
  _Float16* wk  = (_Float16*)alloc((size_t)512 * DD * 2);
  _Float16* wv  = (_Float16*)alloc((size_t)768 * DD * 2);
  _Float16* wo  = (_Float16*)alloc((size_t)768 * DD * 2);
  _Float16* wr  = (_Float16*)alloc((size_t)512 * NPFc * 2);
  _Float16* qwh = (_Float16*)alloc((size_t)M * 512 * 2);
  _Float16* qrh = (_Float16*)alloc((size_t)M * 512 * 2);
  _Float16* kh  = (_Float16*)alloc((size_t)M * 512 * 2);
  _Float16* vt  = (_Float16*)alloc((size_t)M * 768 * 2);
  _Float16* peh = (_Float16*)alloc((size_t)PPAD * NPFc * 2);
  float*    gb  = (float*)alloc((size_t)PPAD * NBc * 4);
  unsigned* gm  = (unsigned*)alloc(256);
  _Float16* rkh = (_Float16*)alloc((size_t)HH * PPAD * DQKc * 2);
  _Float16* ah  = (_Float16*)alloc((size_t)M * 768 * 2);

  const float qscale = 0.125f;  // dqk^-0.5 folded into Q_w

  // --- stage f16 copies (weights transposed so k-dim is contiguous) ---
  cvt_f16_kernel<<<(M * DD + 255) / 256, 256, 0, stream>>>(x, xh, M * DD);
  cvt_f16_t_kernel<<<(DD * 512 + 255) / 256, 256, 0, stream>>>(Q_w, wq, DD, 512,
                                                               qscale);
  cvt_f16_t_kernel<<<(DD * 512 + 255) / 256, 256, 0, stream>>>(K_w, wk, DD, 512,
                                                               1.f);
  cvt_f16_t_kernel<<<(DD * 768 + 255) / 256, 256, 0, stream>>>(V_w, wv, DD, 768,
                                                               1.f);
  cvt_f16_t_kernel<<<(DD * 768 + 255) / 256, 256, 0, stream>>>(out_w, wo, DD,
                                                               768, 1.f);
  cvt_f16_t_kernel<<<(NPFc * 512 + 255) / 256, 256, 0, stream>>>(relKw, wr,
                                                                 NPFc, 512, 1.f);
  // --- positional features -> pe (PPAD x 96 f16, pad row zero) ---
  init_gmax_kernel<<<1, 32, 0, stream>>>(gm);
  gamma_unnorm_kernel<<<(PP * NBc + 255) / 256, 256, 0, stream>>>(gb, gm);
  pe_assemble_kernel<<<(PPAD * 48 + 255) / 256, 256, 0, stream>>>(gb, gm, peh);

  // --- r_K = pe @ rel_K_w  -> [H][PPAD][DQK] f16 ---
  gemm_wmma_kernel<<<dim3(512 / 256, PPAD / 32), 256, 0, stream>>>(
      peh, wr, rkh, nullptr, nullptr, nullptr, PPAD, 512, NPFc, PPAD, DQKc, 1);
  // --- Q (dual store with r_w_bias / r_r_bias), K, V projections ---
  gemm_wmma_kernel<<<dim3(512 / 256, M / 32), 256, 0, stream>>>(
      xh, wq, qwh, qrh, rwb, rrb, M, 512, DD, LL, DQKc, 4);
  gemm_wmma_kernel<<<dim3(512 / 256, M / 32), 256, 0, stream>>>(
      xh, wk, kh, nullptr, nullptr, nullptr, M, 512, DD, LL, DQKc, 1);
  gemm_wmma_kernel<<<dim3(768 / 256, M / 32), 256, 0, stream>>>(
      xh, wv, vt, nullptr, nullptr, nullptr, M, 768, DD, LL, DVc, 2);

  // --- attention (flash-style, logits never materialized) ---
  attn_wmma_kernel<<<dim3(LL / 64, BB * HH), 128, 0, stream>>>(qwh, qrh, kh, vt,
                                                               rkh, ah);
  // --- output projection + bias -> f32 d_out ---
  gemm_wmma_kernel<<<dim3(768 / 256, M / 32), 256, 0, stream>>>(
      ah, wo, d_out, nullptr, out_b, nullptr, M, 768, DD, LL, DVc, 3);
}
